// SinkAwareAttention_44873818308818
// MI455X (gfx1250) — matrix operand order repaired and checked by reading, not compiled, not run on hardware
//
#include <hip/hip_runtime.h>
#include <hip/hip_bf16.h>

// Problem constants (from reference)
#define T_SEQ   4096
#define DM      1024
#define NHEADS  16
#define DHEAD   64
#define NSINKS  4
#define WINDOW  512
#define BATCH   4
#define MROWS   (BATCH * T_SEQ)          // 16384
#define T_CTX   (NSINKS + WINDOW)        // 516
#define Q_FIRST (T_SEQ - T_CTX)          // 3580: first row with any unmasked col

typedef __bf16 bf16_t;
typedef bf16_t v16bf __attribute__((ext_vector_type(16)));
typedef float  v8f   __attribute__((ext_vector_type(8)));
typedef unsigned int v4u __attribute__((ext_vector_type(4)));
typedef unsigned int u32x4 __attribute__((ext_vector_type(4)));
typedef unsigned int u32x8 __attribute__((ext_vector_type(8)));

// ---------------------------------------------------------------------------
// fp32 -> bf16 conversion (grid-stride)
// ---------------------------------------------------------------------------
__global__ void __launch_bounds__(256)
cvt_f32_bf16(const float* __restrict__ src, bf16_t* __restrict__ dst, int n) {
    int i = blockIdx.x * blockDim.x + threadIdx.x;
    int stride = gridDim.x * blockDim.x;
    for (; i < n; i += stride) dst[i] = (bf16_t)src[i];
}

__global__ void __launch_bounds__(256)
zero_u32(unsigned* __restrict__ p, int n) {
    int i = blockIdx.x * blockDim.x + threadIdx.x;
    int stride = gridDim.x * blockDim.x;
    for (; i < n; i += stride) p[i] = 0u;
}

// ---------------------------------------------------------------------------
// bf16 WMMA GEMM: C[M,N] = A[M,K] * B[K,N], f32 accumulate.
// Block: 256 threads = 8 waves. Block tile 128x128, wave tile 32x64, K step 32.
//   A tile (128x32, row-major) : TDM tensor_load_to_lds, double-buffered so the
//                                DMA of tile k0+32 overlaps WMMA on tile k0.
//   B tile (32x128)            : staged transposed in padded LDS so each
//                                fragment is 2x aligned ds_load_b128.
// ---------------------------------------------------------------------------
constexpr int BM = 128, BN = 128, BK = 32;
constexpr int LDB = BK + 8;   // padded K-stride (elems) in transposed LDS B tile

// Issue a 2D TDM load: tile [BK x BM] (elems x rows) of bf16 from gaddr into
// LDS offset ldsOff. tensor_dim0=K (row length), tensor_dim1=M, stride=K.
__device__ __forceinline__ void tdm_load_A(unsigned long long gaddr,
                                           unsigned ldsOff, int M, int K)
{
    const u32x4 g0 = {
        1u,                                   // count=1, user descriptor
        ldsOff,                               // lds_addr
        (unsigned)gaddr,                      // global_addr[31:0]
        ((unsigned)(gaddr >> 32) & 0x01FFFFFFu) | (2u << 30)  // addr hi | type=2
    };
    const u32x8 g1 = {
        0x00010000u,                                    // data_size=1 (2 bytes)
        ((unsigned)K & 0xFFFFu) << 16,                  // tensor_dim0 lo
        (((unsigned)K >> 16) & 0xFFFFu) | (((unsigned)M & 0xFFFFu) << 16), // dim0 hi | dim1 lo
        (((unsigned)M >> 16) & 0xFFFFu) | ((unsigned)BK << 16),            // dim1 hi | tile_dim0
        (unsigned)BM,                                   // tile_dim1 | tile_dim2=0
        (unsigned)K,                                    // tensor_dim0_stride lo
        0u, 0u
    };
    asm volatile("tensor_load_to_lds %0, %1" :: "s"(g0), "s"(g1) : "memory");
}

template <bool OUTF32>
__global__ void __launch_bounds__(256)
gemm_bf16_wmma(const bf16_t* __restrict__ A, const bf16_t* __restrict__ B,
               void* __restrict__ Cout, int M, int N, int K)
{
    __shared__ alignas(16) bf16_t ldsA[2][BM * BK];   // 2 x 8 KB, row-major
    __shared__ alignas(16) bf16_t ldsB[BN * LDB];     // 10 KB, transposed+pad

    const int tid  = threadIdx.x;
    const int lane = tid & 31;
    const int wave = tid >> 5;
    const int wy   = wave >> 1;        // 0..3 : 32-row strip
    const int wx   = wave & 1;         // 0..1 : 64-col strip
    const int nb   = blockIdx.x * BN;
    const int mb   = blockIdx.y * BM;
    const int lm   = lane & 15;        // matrix row (A) / col (B) within tile
    const int lhi  = lane >> 4;        // half-wave selector

    v8f acc[2][4] = {};

    // staging assignment: thread -> (k-row, 16-col chunk) of the 32x128 B tile
    const int skk = tid >> 3;          // 0..31
    const int snn = (tid & 7) * 16;    // 0,16,...,112

    const unsigned long long gaBase =
        (unsigned long long)(const void*)(A + (size_t)mb * K);
    const unsigned ldsA0 = (unsigned)(unsigned long long)(size_t)(&ldsA[0][0]);
    const unsigned ldsA1 = (unsigned)(unsigned long long)(size_t)(&ldsA[1][0]);

    // preload first A tile via TDM (wave 0 issues; EXEC-independent DMA)
    if (wave == 0)
        tdm_load_A(gaBase, ldsA0, M, K);

    const int akoff = lhi * 8;         // A fragment: K halves at +0 / +16
    int ib = 0;

    for (int k0 = 0; k0 < K; k0 += BK, ib ^= 1) {
        { // ---- stage B tile (transposed) into LDS ----
            const bf16_t* gsrc = B + (size_t)(k0 + skk) * N + nb + snn;
            union { v4u u[2]; bf16_t h[16]; } bld;
            bld.u[0] = *(const v4u*)gsrc;
            bld.u[1] = *(const v4u*)(gsrc + 8);
            if (k0 + BK < K)   // prefetch next K tile -> global_prefetch_b8
                __builtin_prefetch(gsrc + (size_t)BK * N, 0, 1);
            #pragma unroll
            for (int j = 0; j < 16; ++j)
                ldsB[(snn + j) * LDB + skk] = bld.h[j];
        }

        // ---- wave 0: kick next A-tile DMA, then wait for current tile ----
        if (wave == 0) {
            if (k0 + BK < K) {
                tdm_load_A(gaBase + (size_t)(k0 + BK) * 2,
                           ib ? ldsA0 : ldsA1, M, K);
                __builtin_amdgcn_s_wait_tensorcnt(1);  // current done, next in flight
            } else {
                __builtin_amdgcn_s_wait_tensorcnt(0);
            }
        }
        __syncthreads();

        // ---- A fragments from LDS: row lm(+16*mi+32*wy), K halves akoff/+16 ----
        const bf16_t* aTile = &ldsA[ib][0];
        v16bf afrag[2];
        #pragma unroll
        for (int mi = 0; mi < 2; ++mi) {
            const bf16_t* ap = aTile + (wy * 32 + mi * 16 + lm) * BK + akoff;
            union { v4u u[2]; v16bf v; } au;
            au.u[0] = *(const v4u*)ap;
            au.u[1] = *(const v4u*)(ap + 16);
            afrag[mi] = au.v;
        }

        // ---- B fragments from transposed LDS + WMMA ----
        #pragma unroll
        for (int ni = 0; ni < 4; ++ni) {
            const bf16_t* bp = &ldsB[(wx * 64 + ni * 16 + lm) * LDB + (lhi << 3)];
            union { v4u u[2]; v16bf v; } bu;
            bu.u[0] = *(const v4u*)bp;
            bu.u[1] = *(const v4u*)(bp + 16);
            const v16bf bfrag = bu.v;
            #pragma unroll
            for (int mi = 0; mi < 2; ++mi) {
                acc[mi][ni] = __builtin_amdgcn_wmma_f32_16x16x32_bf16(
                    false, afrag[mi], false, bfrag,
                    (short)0, acc[mi][ni], false, false);
            }
        }
        __syncthreads();   // everyone done reading before restage/DMA overwrite
    }

    // ---- store: VGPR r holds rows r (lanes 0-15) / r+8 (lanes 16-31), col lm ----
    #pragma unroll
    for (int mi = 0; mi < 2; ++mi) {
        const int r0 = mb + wy * 32 + mi * 16 + lhi * 8;
        #pragma unroll
        for (int ni = 0; ni < 4; ++ni) {
            const int col = nb + wx * 64 + ni * 16 + lm;
            #pragma unroll
            for (int r = 0; r < 8; ++r) {
                const size_t idx = (size_t)(r0 + r) * N + col;
                if constexpr (OUTF32) ((float*)Cout)[idx] = acc[mi][ni][r];
                else                  ((bf16_t*)Cout)[idx] = (bf16_t)acc[mi][ni][r];
            }
        }
    }
}

// ---------------------------------------------------------------------------
// Sink-aware attention, exploiting that rows < Q_FIRST are fully masked (the
// reference's NaN-softmax -> zeros path). One wave32 per live query row;
// each lane owns 2 of the 64 head dims; online softmax over <=516 context
// keys (cols 0..3 = sinks at positions 0..3, col c>=4 = position 3580+c).
// ---------------------------------------------------------------------------
__global__ void __launch_bounds__(128)
attn_kernel(const bf16_t* __restrict__ Q, const bf16_t* __restrict__ Kb,
            const bf16_t* __restrict__ Vb, bf16_t* __restrict__ O)
{
    const int lane = threadIdx.x & 31;
    const int wave = threadIdx.x >> 5;
    const int q = Q_FIRST + blockIdx.x * 4 + wave;
    if (q >= T_SEQ) return;
    const int bh = blockIdx.y;
    const int b  = bh >> 4;
    const int h  = bh & 15;

    const size_t rowQ = ((size_t)(b * T_SEQ + q)) * DM + h * DHEAD;
    const float q0 = (float)Q[rowQ + 2 * lane];
    const float q1 = (float)Q[rowQ + 2 * lane + 1];

    float m = -3.0e38f, lsum = 0.f, a0 = 0.f, a1 = 0.f;
    const int ncols = q - Q_FIRST + 1;   // visible cols: 0..q-3580

    for (int c = 0; c < ncols; ++c) {
        const int p = (c < NSINKS) ? c : (Q_FIRST + c);  // key/value position
        const size_t rk = ((size_t)(b * T_SEQ + p)) * DM + h * DHEAD;
        float partial = q0 * (float)Kb[rk + 2 * lane]
                      + q1 * (float)Kb[rk + 2 * lane + 1];
        #pragma unroll
        for (int off = 16; off; off >>= 1)
            partial += __shfl_xor(partial, off, 32);
        const float logit = partial * 0.125f;            // 1/sqrt(64)

        const float mn   = fmaxf(m, logit);
        const float corr = __expf(m - mn);
        const float w    = __expf(logit - mn);
        m = mn;
        lsum = lsum * corr + w;
        a0 = a0 * corr + w * (float)Vb[rk + 2 * lane];
        a1 = a1 * corr + w * (float)Vb[rk + 2 * lane + 1];
    }

    const float inv = 1.0f / lsum;
    O[rowQ + 2 * lane]     = (bf16_t)(a0 * inv);
    O[rowQ + 2 * lane + 1] = (bf16_t)(a1 * inv);
}

// ---------------------------------------------------------------------------
extern "C" void kernel_launch(void* const* d_in, const int* in_sizes, int n_in,
                              void* d_out, int out_size, void* d_ws, size_t ws_size,
                              hipStream_t stream)
{
    const float* x  = (const float*)d_in[0];
    const float* Wq = (const float*)d_in[1];
    const float* Wk = (const float*)d_in[2];
    const float* Wv = (const float*)d_in[3];
    const float* Wo = (const float*)d_in[4];

    char* ws = (char*)d_ws;
    size_t off = 0;
    auto take = [&](size_t bytes) -> char* {
        char* p = ws + off;
        off += (bytes + 255) & ~(size_t)255;
        return p;
    };

    const size_t nX = (size_t)MROWS * DM;   // 16M elems
    const size_t nW = (size_t)DM * DM;      // 1M elems

    bf16_t* xb  = (bf16_t*)take(nX * 2);
    bf16_t* wqb = (bf16_t*)take(nW * 2);
    bf16_t* wkb = (bf16_t*)take(nW * 2);
    bf16_t* wvb = (bf16_t*)take(nW * 2);
    bf16_t* wob = (bf16_t*)take(nW * 2);
    bf16_t* Qb  = (bf16_t*)take(nX * 2);
    bf16_t* Kb  = (bf16_t*)take(nX * 2);
    bf16_t* Vb  = (bf16_t*)take(nX * 2);
    bf16_t* Ab  = (bf16_t*)take(nX * 2);    // attention output (bf16)

    // 1) convert inputs to bf16
    cvt_f32_bf16<<<2048, 256, 0, stream>>>(x,  xb,  (int)nX);
    cvt_f32_bf16<<<512,  256, 0, stream>>>(Wq, wqb, (int)nW);
    cvt_f32_bf16<<<512,  256, 0, stream>>>(Wk, wkb, (int)nW);
    cvt_f32_bf16<<<512,  256, 0, stream>>>(Wv, wvb, (int)nW);
    cvt_f32_bf16<<<512,  256, 0, stream>>>(Wo, wob, (int)nW);

    // 2) Q/K/V projections (bf16 WMMA, bf16 out)
    dim3 gg(DM / BN, MROWS / BM);   // (8, 128)
    gemm_bf16_wmma<false><<<gg, 256, 0, stream>>>(xb, wqb, Qb, MROWS, DM, DM);
    gemm_bf16_wmma<false><<<gg, 256, 0, stream>>>(xb, wkb, Kb, MROWS, DM, DM);
    gemm_bf16_wmma<false><<<gg, 256, 0, stream>>>(xb, wvb, Vb, MROWS, DM, DM);

    // 3) attention (zero dead rows, fill the 516 live rows per (b,h))
    zero_u32<<<2048, 256, 0, stream>>>((unsigned*)Ab, (int)(nX / 2));
    dim3 ga((T_CTX + 3) / 4, BATCH * NHEADS);   // (129, 64)
    attn_kernel<<<ga, 128, 0, stream>>>(Qb, Kb, Vb, Ab);

    // 4) output projection, f32 result straight into d_out
    gemm_bf16_wmma<true><<<gg, 256, 0, stream>>>(Ab, wob, d_out, MROWS, DM, DM);
}
